// WindowMultiHeadAttention_68719477580
// MI455X (gfx1250) — matrix operand order repaired and checked by reading, not compiled, not run on hardware
//
#include <hip/hip_runtime.h>
#include <hip/hip_bf16.h>

typedef __attribute__((ext_vector_type(16))) _Float16 v16h;
typedef __attribute__((ext_vector_type(8)))  _Float16 v8h;
typedef __attribute__((ext_vector_type(8)))  float    v8f;

#define EMB  384
#define NH   12
#define HD   32
#define LW   49
#define LP   64
#define NWIN 64

static __device__ __forceinline__ v16h cat8(v8h lo, v8h hi) {
  return __builtin_shufflevector(lo, hi, 0,1,2,3,4,5,6,7,8,9,10,11,12,13,14,15);
}

// A fragment (16x32 f16) of a row-major matrix (stride in halves).
// lane l: row m0+(l&15), 16 consecutive K at k0+16*(l>=16).
static __device__ __forceinline__ v16h ld_afrag(const _Float16* base, int stride,
                                                int m0, int k0, int lane) {
  const _Float16* p = base + (m0 + (lane & 15)) * stride + k0 + ((lane >> 4) << 4);
  v8h lo = *(const v8h*)p;
  v8h hi = *(const v8h*)(p + 8);
  return cat8(lo, hi);
}

// B fragment (32x16 f16): lane holds 16 consecutive K of its column; caller
// passes the per-lane pointer into an N-major (transposed) matrix.
static __device__ __forceinline__ v16h ld_bfrag(const _Float16* p) {
  v8h lo = *(const v8h*)p;
  v8h hi = *(const v8h*)(p + 8);
  return cat8(lo, hi);
}

static __device__ __forceinline__ v8f wmma16(v16h a, v16h b, v8f c) {
  return __builtin_amdgcn_wmma_f32_16x16x32_f16(false, a, false, b, (short)0, c,
                                                false, false);
}

// ---- prep: fp32 weight [K][N] -> f16 transposed [N][K] ----
__global__ void prep_wT(const float* __restrict__ w, _Float16* __restrict__ wT,
                        int K, int N) {
  int idx = blockIdx.x * blockDim.x + threadIdx.x;
  if (idx >= K * N) return;
  int n = idx / K;
  int k = idx - n * K;
  wT[idx] = (_Float16)w[(size_t)k * N + n];
}

// ---- prep: gathered relative position bias [H][L][L] fp32 ----
__global__ void prep_bias(const float* __restrict__ table, const int* __restrict__ ridx,
                          float* __restrict__ biasG) {
  int t = blockIdx.x * blockDim.x + threadIdx.x;
  if (t >= NH * LW * LW) return;
  int h = t / (LW * LW);
  int rem = t - h * (LW * LW);
  biasG[t] = table[ridx[rem] * NH + h];
}

// ---- fused window attention: one block (8 waves) per (b, window) ----
__global__ __launch_bounds__(256) void wmha_main(
    const float* __restrict__ x, const float* __restrict__ mask,
    const float* __restrict__ b_qkv, const float* __restrict__ b_out,
    const _Float16* __restrict__ wqkvT, const _Float16* __restrict__ woutT,
    const float* __restrict__ biasG, float* __restrict__ out) {
  __shared__ __align__(16) _Float16 Xs[LP * 392];  // x tile, f16, padded stride
  __shared__ __align__(16) _Float16 Cx[LP * 392];  // context (all heads), f16
  __shared__ __align__(16) _Float16 Qh[LP * 40];   // per-head Q*scale 64x32
  __shared__ __align__(16) _Float16 Kh[LP * 40];   // per-head K 64x32 (row-major = K^T N-major)
  __shared__ __align__(16) _Float16 Vt[HD * 72];   // per-head V transposed 32x64
  __shared__ __align__(16) float    Sc[LP * 68];   // scaled scores fp32 64x64
  __shared__ __align__(16) _Float16 Al[LP * 72];   // alpha f16 64x64

  const int tid  = threadIdx.x;
  const int lane = tid & 31;
  const int wv   = tid >> 5;
  const int blk  = blockIdx.x;
  const int win  = blk & (NWIN - 1);

  const float* xb = x   + (size_t)blk * LW * EMB;
  float*       ob = out + (size_t)blk * LW * EMB;

  // load + convert x tile (rows >= 49 zero-padded)
  for (int r = wv; r < LP; r += 8)
    for (int c = lane; c < EMB; c += 32)
      Xs[r * 392 + c] = (r < LW) ? (_Float16)xb[r * EMB + c] : (_Float16)0.0f;
  __syncthreads();

  const int hi8  = (lane >> 4) << 3;  // 0 or 8 (row offset for C rows)
  const int colr = lane & 15;
  const float scale = 0.17677669529663687f;  // 1/sqrt(32), folded into Q

  for (int h = 0; h < NH; ++h) {
    // ---- QKV head slice: 64 x 96 (q|k|v) = 24 tiles, 3 per wave ----
    for (int t = wv * 3; t < wv * 3 + 3; ++t) {
      const int mt = t & 3, nt = t >> 2;
      const int m0 = mt << 4;
      const int sec = nt >> 1;            // 0=q 1=k 2=v
      const int c0  = (nt & 1) << 4;
      const int gcol = sec * EMB + h * HD + c0 + colr;
      const _Float16* bp = wqkvT + (size_t)gcol * EMB + ((lane >> 4) << 4);
      v8f acc = {};
      for (int kk = 0; kk < EMB; kk += 32) {
        v16h a = ld_afrag(Xs, 392, m0, kk, lane);
        v16h b = ld_bfrag(bp + kk);
        acc = wmma16(a, b, acc);
      }
      const float bias = b_qkv[gcol];
      const int cl = c0 + colr;
      if (sec == 0) {
#pragma unroll
        for (int r = 0; r < 8; ++r)
          Qh[(m0 + r + hi8) * 40 + cl] = (_Float16)((acc[r] + bias) * scale);
      } else if (sec == 1) {
#pragma unroll
        for (int r = 0; r < 8; ++r)
          Kh[(m0 + r + hi8) * 40 + cl] = (_Float16)(acc[r] + bias);
      } else {
#pragma unroll
        for (int r = 0; r < 8; ++r)
          Vt[cl * 72 + (m0 + r + hi8)] = (_Float16)(acc[r] + bias);
      }
    }
    __syncthreads();

    // ---- scores = (Q*scale) @ K^T : 16 tiles, 2 per wave ----
    for (int t = wv * 2; t < wv * 2 + 2; ++t) {
      const int mt = t & 3, nt = t >> 2;
      const int m0 = mt << 4, n0 = nt << 4;
      v16h a = ld_afrag(Qh, 40, m0, 0, lane);
      v16h b = ld_bfrag(Kh + (n0 + colr) * 40 + ((lane >> 4) << 4));
      v8f acc = {};
      acc = wmma16(a, b, acc);
#pragma unroll
      for (int r = 0; r < 8; ++r)
        Sc[(m0 + r + hi8) * 68 + n0 + colr] = acc[r];
    }
    __syncthreads();

    // ---- softmax (+rel-bias +mask): one wave per row, shfl reductions ----
    for (int i = wv; i < LP; i += 8) {
      if (i < LW) {
        const float* bh = biasG + (h * LW + i) * LW;
        const float* mr = mask + ((size_t)win * LW + i) * LW;
        const int j0 = lane, j1 = lane + 32;
        float v0 = Sc[i * 68 + j0] + bh[j0] + mr[j0];          // j0 < 32 < 49 always
        float v1 = (j1 < LW) ? (Sc[i * 68 + j1] + bh[j1] + mr[j1]) : -1e30f;
        float m = fmaxf(v0, v1);
#pragma unroll
        for (int off = 16; off > 0; off >>= 1) m = fmaxf(m, __shfl_xor(m, off, 32));
        float e0 = __expf(v0 - m);
        float e1 = (j1 < LW) ? __expf(v1 - m) : 0.0f;
        float s = e0 + e1;
#pragma unroll
        for (int off = 16; off > 0; off >>= 1) s += __shfl_xor(s, off, 32);
        const float inv = 1.0f / s;
        Al[i * 72 + j0] = (_Float16)(e0 * inv);
        Al[i * 72 + j1] = (_Float16)(e1 * inv);
      } else {
        Al[i * 72 + lane]      = (_Float16)0.0f;
        Al[i * 72 + lane + 32] = (_Float16)0.0f;
      }
    }
    __syncthreads();

    // ---- context = alpha @ V : 8 tiles, 1 per wave (K=64 -> 2 WMMA) ----
    {
      const int mt = wv & 3, nt = wv >> 2;
      const int m0 = mt << 4, n0 = nt << 4;
      v8f acc = {};
#pragma unroll
      for (int kk = 0; kk < 64; kk += 32) {
        v16h a = ld_afrag(Al, 72, m0, kk, lane);
        v16h b = ld_bfrag(Vt + (n0 + colr) * 72 + kk + ((lane >> 4) << 4));
        acc = wmma16(a, b, acc);
      }
      const int cl = h * HD + n0 + colr;
#pragma unroll
      for (int r = 0; r < 8; ++r)
        Cx[(m0 + r + hi8) * 392 + cl] = (_Float16)acc[r];
    }
    __syncthreads();
  }

  // ---- output projection: 96 tiles, 12 per wave ----
  for (int t = wv * 12; t < wv * 12 + 12; ++t) {
    const int mt = t & 3, nt = t >> 2;
    const int m0 = mt << 4, n0 = nt << 4;
    const int gcol = n0 + colr;
    const _Float16* bp = woutT + (size_t)gcol * EMB + ((lane >> 4) << 4);
    v8f acc = {};
    for (int kk = 0; kk < EMB; kk += 32) {
      v16h a = ld_afrag(Cx, 392, m0, kk, lane);
      v16h b = ld_bfrag(bp + kk);
      acc = wmma16(a, b, acc);
    }
    const float bo = b_out[gcol];
#pragma unroll
    for (int r = 0; r < 8; ++r) {
      const int row = m0 + r + hi8;
      if (row < LW) ob[row * EMB + gcol] = acc[r] + bo;
    }
  }
}

extern "C" void kernel_launch(void* const* d_in, const int* in_sizes, int n_in,
                              void* d_out, int out_size, void* d_ws, size_t ws_size,
                              hipStream_t stream) {
  (void)in_sizes; (void)n_in; (void)out_size; (void)ws_size;
  const float* x      = (const float*)d_in[0];
  const float* mask   = (const float*)d_in[1];
  const float* w_qkv  = (const float*)d_in[2];
  const float* b_qkv  = (const float*)d_in[3];
  const float* w_out  = (const float*)d_in[4];
  const float* b_out  = (const float*)d_in[5];
  const float* table  = (const float*)d_in[6];
  const int*   relidx = (const int*)d_in[7];
  float* out = (float*)d_out;

  char* ws = (char*)d_ws;
  _Float16* wqkvT = (_Float16*)ws;                       // 1152*384*2 = 884736 B
  _Float16* woutT = (_Float16*)(ws + 884736);            //  384*384*2 = 294912 B
  float*    biasG = (float*)(ws + 884736 + 294912);      // 12*49*49*4 = 115248 B

  prep_wT<<<(EMB * 3 * EMB + 255) / 256, 256, 0, stream>>>(w_qkv, wqkvT, EMB, 3 * EMB);
  prep_wT<<<(EMB * EMB + 255) / 256, 256, 0, stream>>>(w_out, woutT, EMB, EMB);
  prep_bias<<<(NH * LW * LW + 255) / 256, 256, 0, stream>>>(table, relidx, biasG);
  wmha_main<<<64 * NWIN, 256, 0, stream>>>(x, mask, b_qkv, b_out, wqkvT, woutT,
                                           biasG, out);
}
